// MultiHeadAttention_10668698763481
// MI455X (gfx1250) — matrix operand order repaired and checked by reading, not compiled, not run on hardware
//
#include <hip/hip_runtime.h>
#include <hip/hip_bf16.h>

typedef __attribute__((ext_vector_type(16))) _Float16 v16h;
typedef __attribute__((ext_vector_type(8)))  _Float16 v8h;
typedef __attribute__((ext_vector_type(4)))  _Float16 v4h;
typedef __attribute__((ext_vector_type(8)))  float    v8f;
typedef __attribute__((ext_vector_type(4)))  int      v4i;

#define DM   1024
#define TOK  8192      // 8 * 1024 tokens
#define NH   16
#define HD   64

#define AS_GLOBAL __attribute__((address_space(1)))
#define AS_LDS    __attribute__((address_space(3)))

// ---- async global->LDS copy (gfx1250 GLOBAL_LOAD_ASYNC_TO_LDS_B128, ASYNCcnt) ----
#if defined(__gfx1250__) && __has_builtin(__builtin_amdgcn_global_load_async_to_lds_b128)
#define ASYNC_CP_B128(dst, src)                                              \
    __builtin_amdgcn_global_load_async_to_lds_b128(                          \
        (v4i AS_GLOBAL*)(src), (v4i AS_LDS*)(dst), 0, 0)
#else
#define ASYNC_CP_B128(dst, src) (*(v8h*)(dst) = *(const v8h*)(src))
#endif
#define ASYNC_WAIT() asm volatile("s_wait_asynccnt 0x0" ::: "memory")

__device__ __forceinline__ v8f wmma16(v16h a, v16h b, v8f c) {
    return __builtin_amdgcn_wmma_f32_16x16x32_f16(false, a, false, b, (short)0, c,
                                                  false, false);
}

// Build a 16-half A fragment from two contiguous 8-half chunks
__device__ __forceinline__ v16h make_afrag(const _Float16* p0, const _Float16* p1) {
    v8h lo = *(const v8h*)p0;
    v8h hi = *(const v8h*)p1;
    v16h a;
#pragma unroll
    for (int i = 0; i < 8; ++i) { a[i] = lo[i]; a[i + 8] = hi[i]; }
    return a;
}

// ---------------------------------------------------------------- cast weights
__global__ void cast_f16_kernel(const float* __restrict__ x,
                                _Float16* __restrict__ y, int n) {
    int i = blockIdx.x * blockDim.x + threadIdx.x;
    if (i < n) y[i] = (_Float16)x[i];
}

// ---------------------------------------------------------------- layernorm
__global__ void ln_f16_kernel(const float* __restrict__ x,
                              const float* __restrict__ gamma,
                              const float* __restrict__ beta,
                              _Float16* __restrict__ y) {
    __shared__ float red[256];
    const int row = blockIdx.x;
    const int tid = threadIdx.x;
    const float* xr = x + (size_t)row * DM;
    float4 v = ((const float4*)xr)[tid];

    float s = v.x + v.y + v.z + v.w;
    red[tid] = s;
    __syncthreads();
#pragma unroll
    for (int off = 128; off > 0; off >>= 1) {
        if (tid < off) red[tid] += red[tid + off];
        __syncthreads();
    }
    float mu = red[0] * (1.0f / DM);
    __syncthreads();

    float dx = v.x - mu, dy = v.y - mu, dz = v.z - mu, dw = v.w - mu;
    red[tid] = dx * dx + dy * dy + dz * dz + dw * dw;
    __syncthreads();
#pragma unroll
    for (int off = 128; off > 0; off >>= 1) {
        if (tid < off) red[tid] += red[tid + off];
        __syncthreads();
    }
    float rstd = rsqrtf(red[0] * (1.0f / DM) + 1e-5f);

    int c = tid * 4;
    v4h o;
    o[0] = (_Float16)(dx * rstd * gamma[c + 0] + beta[c + 0]);
    o[1] = (_Float16)(dy * rstd * gamma[c + 1] + beta[c + 1]);
    o[2] = (_Float16)(dz * rstd * gamma[c + 2] + beta[c + 2]);
    o[3] = (_Float16)(dw * rstd * gamma[c + 3] + beta[c + 3]);
    *((v4h*)(y + (size_t)row * DM) + tid) = o;
}

// ---------------------------------------------------------------- projection GEMM
// C[M=8192, N=1024] = A[8192,1024] * W[1024,1024]^T + bias, stored f16.
// transpose_v: store as Vt[b][n][l] (for P*V B-fragments) instead of [token][n].
__global__ void gemm_f16_kernel(const _Float16* __restrict__ A,
                                const _Float16* __restrict__ W,
                                const float* __restrict__ bias,
                                _Float16* __restrict__ C, int transpose_v) {
    const int lane = threadIdx.x & 31;
    const int wave = blockIdx.x * (blockDim.x >> 5) + (threadIdx.x >> 5);
    const int tm   = wave >> 4;          // 0..511  (M tile)
    const int tn0  = (wave & 15) * 4;    // 4 consecutive N tiles per wave
    const int half = lane >> 4, lr = lane & 15;

    const _Float16* arow = A + (size_t)(tm * 16 + lr) * DM;
    v8f acc[4] = {};

    for (int kk = 0; kk < DM; kk += 32) {
        const int ka = kk + half * 16;
        v16h a = make_afrag(arow + ka, arow + ka + 8);
        const int kb = kk + half * 16;
#pragma unroll
        for (int t = 0; t < 4; ++t) {
            const _Float16* wrow = W + (size_t)((tn0 + t) * 16 + lr) * DM + kb;
            v16h b = *(const v16h*)wrow;
            acc[t] = wmma16(a, b, acc[t]);
        }
    }

#pragma unroll
    for (int t = 0; t < 4; ++t) {
        const int col = (tn0 + t) * 16 + lr;
        const float bv = bias[col];
#pragma unroll
        for (int r = 0; r < 8; ++r) {
            const int row = tm * 16 + r + half * 8;   // global token
            _Float16 val = (_Float16)(acc[t][r] + bv);
            if (!transpose_v) {
                C[(size_t)row * DM + col] = val;
            } else {
                const int bb = row >> 10, l = row & 1023;
                C[((size_t)bb * DM + col) * 1024 + l] = val;
            }
        }
    }
}

// ---------------------------------------------------------------- flash attention
// Block = 8 waves sharing one (b, h); wave wv handles q-tile qg*8+wv.
// K/V tiles (32 keys x 64 dims) are staged into double-buffered LDS with async
// global->LDS copies (ASYNCcnt): next tile's copy overlaps current tile's
// 8 WMMAs + softmax. One barrier per 32-key step.
// Qf/Kf: [b,l,1024] f16, Vt: [b, h*64+d, l] f16.
__global__ void attention_kernel(const _Float16* __restrict__ Qf,
                                 const _Float16* __restrict__ Kf,
                                 const _Float16* __restrict__ Vt,
                                 const unsigned char* __restrict__ mask,
                                 _Float16* __restrict__ att) {
    __shared__ __align__(32) _Float16 ptile[8][16 * 32];
    __shared__ __align__(32) _Float16 kbuf[2][32 * 64];   // [key within 32][d 0..63]
    __shared__ __align__(32) _Float16 vbuf[2][64 * 32];   // [d 0..63][key within 32]

    const int tid  = threadIdx.x;
    const int lane = tid & 31;
    const int wv   = tid >> 5;
    const int qg = blockIdx.x & 7;
    const int h  = (blockIdx.x >> 3) & 15;
    const int b  = blockIdx.x >> 7;
    const int qt = qg * 8 + wv;
    const int half = lane >> 4, lr = lane & 15;

    // Q A-fragments for d-chunks 0..31 and 32..63
    const _Float16* qrow = Qf + ((size_t)(b * 1024 + qt * 16 + lr)) * DM + h * HD;
    v16h aq0 = make_afrag(qrow + half * 16,      qrow + half * 16 + 8);
    v16h aq1 = make_afrag(qrow + 32 + half * 16, qrow + 32 + half * 16 + 8);

    const _Float16* kbase = Kf + ((size_t)b * 1024) * DM + h * HD;       // + j*DM
    const _Float16* vbase = Vt + ((size_t)(b * DM + h * HD)) * 1024;     // + n*1024 + j
    const unsigned char* mrow = mask + b * 1024;

    // cooperative-copy assignment (fixed per thread): 256 threads x 16B each buffer
    const int kkey = tid >> 3, kseg = tid & 7;   // K tile: 32 keys x 128B
    const int vn   = tid >> 2, vseg = tid & 3;   // V tile: 64 dims x 64B
    const _Float16* ksrc0 = kbase + (size_t)kkey * DM + kseg * 8;  // + jj*DM step
    const _Float16* vsrc0 = vbase + (size_t)vn * 1024 + vseg * 8;  // + jj step

    float m[8], lsum[8];
    v8f o[4] = {};
#pragma unroll
    for (int r = 0; r < 8; ++r) { m[r] = -3.0e38f; lsum[r] = 0.0f; }

    _Float16* pt = ptile[wv];

    // prologue: fill buffer 0 with keys [0, 32)
    ASYNC_CP_B128(kbuf[0] + kkey * 64 + kseg * 8, ksrc0);
    ASYNC_CP_B128(vbuf[0] + vn * 32 + vseg * 8, vsrc0);

    for (int i = 0; i < 32; ++i) {
        const int jj  = i * 32;
        const int cur = i & 1;

        ASYNC_WAIT();        // this wave's copies for `cur` have landed
        __syncthreads();     // all waves' copies landed; `cur^1` free to refill

        if (i + 1 < 32) {    // prefetch next tile into the other buffer (no wait)
            const int nxt = cur ^ 1;
            ASYNC_CP_B128(kbuf[nxt] + kkey * 64 + kseg * 8,
                          ksrc0 + (size_t)(jj + 32) * DM);
            ASYNC_CP_B128(vbuf[nxt] + vn * 32 + vseg * 8,
                          vsrc0 + (jj + 32));
        }

        // ---- scores: S(16x32) = Q(16x64) x K^T, from LDS ----
        v8f s[2];
#pragma unroll
        for (int t = 0; t < 2; ++t) {
            const _Float16* krow = kbuf[cur] + (t * 16 + lr) * 64 + half * 16;
            v16h b0 = *(const v16h*)krow;          // d = 0..31 slice
            v16h b1 = *(const v16h*)(krow + 32);   // d = 32..63 slice
            v8f z = {};
            z = wmma16(aq0, b0, z);
            z = wmma16(aq1, b1, z);
            const int j = jj + t * 16 + lr;
            const int masked = mrow[j] ? 1 : 0;
#pragma unroll
            for (int r = 0; r < 8; ++r)
                s[t][r] = masked ? -3.0e38f : z[r] * 0.125f;   // 1/sqrt(64)
        }

        // ---- per-row online softmax (row M = r + 8*half, spread over 16 lanes) ----
#pragma unroll
        for (int r = 0; r < 8; ++r) {
            float v = fmaxf(s[0][r], s[1][r]);
            v = fmaxf(v, __shfl_xor(v, 1, 32));
            v = fmaxf(v, __shfl_xor(v, 2, 32));
            v = fmaxf(v, __shfl_xor(v, 4, 32));
            v = fmaxf(v, __shfl_xor(v, 8, 32));
            const float mn = fmaxf(m[r], v);
            const float sf = __expf(m[r] - mn);
            m[r] = mn;

            const float p0 = __expf(s[0][r] - mn);
            const float p1 = __expf(s[1][r] - mn);
            float rs = p0 + p1;
            rs += __shfl_xor(rs, 1, 32);
            rs += __shfl_xor(rs, 2, 32);
            rs += __shfl_xor(rs, 4, 32);
            rs += __shfl_xor(rs, 8, 32);
            lsum[r] = lsum[r] * sf + rs;

#pragma unroll
            for (int t2 = 0; t2 < 4; ++t2) o[t2][r] *= sf;

            const int row = r + half * 8;
            pt[row * 32 + lr]      = (_Float16)p0;
            pt[row * 32 + 16 + lr] = (_Float16)p1;
        }

        // ---- wave-internal LDS reshape: C-layout f32 -> A-fragment f16 ----
        asm volatile("s_wait_dscnt 0x0" ::: "memory");
        const int kp = half * 16;
        v16h ap = make_afrag(pt + lr * 32 + kp, pt + lr * 32 + kp + 8);

        // ---- O += P(16x32) x V(32x64), V^T from LDS ----
#pragma unroll
        for (int t2 = 0; t2 < 4; ++t2) {
            const _Float16* vrow = vbuf[cur] + (t2 * 16 + lr) * 32 + half * 16;
            v16h bvf = *(const v16h*)vrow;
            o[t2] = wmma16(ap, bvf, o[t2]);
        }
    }

    _Float16* abase = att + ((size_t)(b * 1024 + qt * 16)) * DM + h * HD;
#pragma unroll
    for (int r = 0; r < 8; ++r) {
        const float inv = 1.0f / lsum[r];
        const int row = r + half * 8;
#pragma unroll
        for (int t2 = 0; t2 < 4; ++t2)
            abase[(size_t)row * DM + t2 * 16 + lr] = (_Float16)(o[t2][r] * inv);
    }
}

// ---------------------------------------------------------------- output GEMM + residual
__global__ void gemm_out_kernel(const _Float16* __restrict__ A,
                                const _Float16* __restrict__ W,
                                const float* __restrict__ bias,
                                const float* __restrict__ resid,
                                float* __restrict__ out) {
    const int lane = threadIdx.x & 31;
    const int wave = blockIdx.x * (blockDim.x >> 5) + (threadIdx.x >> 5);
    const int tm   = wave >> 4;
    const int tn0  = (wave & 15) * 4;
    const int half = lane >> 4, lr = lane & 15;

    const _Float16* arow = A + (size_t)(tm * 16 + lr) * DM;
    v8f acc[4] = {};

    for (int kk = 0; kk < DM; kk += 32) {
        const int ka = kk + half * 16;
        v16h a = make_afrag(arow + ka, arow + ka + 8);
        const int kb = kk + half * 16;
#pragma unroll
        for (int t = 0; t < 4; ++t) {
            const _Float16* wrow = W + (size_t)((tn0 + t) * 16 + lr) * DM + kb;
            v16h b = *(const v16h*)wrow;
            acc[t] = wmma16(a, b, acc[t]);
        }
    }

#pragma unroll
    for (int t = 0; t < 4; ++t) {
        const int col = (tn0 + t) * 16 + lr;
        const float bv = bias[col];
#pragma unroll
        for (int r = 0; r < 8; ++r) {
            const int row = tm * 16 + r + half * 8;
            out[(size_t)row * DM + col] = resid[(size_t)row * DM + col] + acc[t][r] + bv;
        }
    }
}

// ---------------------------------------------------------------- launcher
extern "C" void kernel_launch(void* const* d_in, const int* in_sizes, int n_in,
                              void* d_out, int out_size, void* d_ws, size_t ws_size,
                              hipStream_t stream) {
    const float* q     = (const float*)d_in[0];
    const float* k     = (const float*)d_in[1];
    const float* v     = (const float*)d_in[2];
    const float* Wq    = (const float*)d_in[3];
    const float* bq    = (const float*)d_in[4];
    const float* Wk    = (const float*)d_in[5];
    const float* bk    = (const float*)d_in[6];
    const float* Wv    = (const float*)d_in[7];
    const float* bv    = (const float*)d_in[8];
    const float* Wo    = (const float*)d_in[9];
    const float* bo    = (const float*)d_in[10];
    const float* gamma = (const float*)d_in[11];
    const float* beta  = (const float*)d_in[12];
    const unsigned char* mask = (const unsigned char*)d_in[13];
    float* out = (float*)d_out;

    const size_t M1 = 1024u * 1024u;   // elements
    _Float16* ws   = (_Float16*)d_ws;
    _Float16* Wq_h = ws;
    _Float16* Wk_h = ws + 1 * M1;
    _Float16* Wv_h = ws + 2 * M1;
    _Float16* Wo_h = ws + 3 * M1;
    _Float16* LNq  = ws + 4 * M1;      // 8M each (8192 x 1024)
    _Float16* LNk  = ws + 12 * M1;
    _Float16* LNv  = ws + 20 * M1;
    _Float16* Qf   = ws + 28 * M1;
    _Float16* Kf   = ws + 36 * M1;
    _Float16* Vt   = ws + 44 * M1;
    _Float16* Att  = ws + 52 * M1;     // total 60M halves = 120 MB

    cast_f16_kernel<<<4096, 256, 0, stream>>>(Wq, Wq_h, (int)M1);
    cast_f16_kernel<<<4096, 256, 0, stream>>>(Wk, Wk_h, (int)M1);
    cast_f16_kernel<<<4096, 256, 0, stream>>>(Wv, Wv_h, (int)M1);
    cast_f16_kernel<<<4096, 256, 0, stream>>>(Wo, Wo_h, (int)M1);

    ln_f16_kernel<<<TOK, 256, 0, stream>>>(q, gamma, beta, LNq);
    ln_f16_kernel<<<TOK, 256, 0, stream>>>(k, gamma, beta, LNk);
    ln_f16_kernel<<<TOK, 256, 0, stream>>>(v, gamma, beta, LNv);

    gemm_f16_kernel<<<1024, 256, 0, stream>>>(LNq, Wq_h, bq, Qf, 0);
    gemm_f16_kernel<<<1024, 256, 0, stream>>>(LNk, Wk_h, bk, Kf, 0);
    gemm_f16_kernel<<<1024, 256, 0, stream>>>(LNv, Wv_h, bv, Vt, 1);

    attention_kernel<<<1024, 256, 0, stream>>>(Qf, Kf, Vt, mask, Att);

    gemm_out_kernel<<<1024, 256, 0, stream>>>(Att, Wo_h, bo, q, out);
}
